// AdvancedTradingModel_57475252355112
// MI455X (gfx1250) — matrix operand order repaired and checked by reading, not compiled, not run on hardware
//
#include <hip/hip_runtime.h>
#include <hip/hip_bf16.h>

// ---------------------------------------------------------------------------
// Model dims
// ---------------------------------------------------------------------------
#define S_LEN 256
#define BATCH 32
#define HDIM  256
#define PDIM  64
#define NSTATE 128
#define DTRK  16
#define NPCNT 8
#define MROWS 64
#define TLEN_ 64
#define EMBD  128

typedef __bf16 bf16;
typedef __attribute__((ext_vector_type(16))) __bf16 v16bf;
typedef __attribute__((ext_vector_type(8)))  float  v8f;

// ---------------------------------------------------------------------------
// Helpers
// ---------------------------------------------------------------------------
__device__ __forceinline__ bf16 f2bf(float f) {
  union { float f; unsigned u; } x; x.f = f;
  unsigned r = (x.u + 0x7FFFu + ((x.u >> 16) & 1u)) >> 16;
  unsigned short h = (unsigned short)r;
  bf16 b; __builtin_memcpy(&b, &h, sizeof(h));
  return b;
}
__device__ __forceinline__ float sigmoidf_(float x) { return 1.0f / (1.0f + __expf(-x)); }

// Opaque zero (SGPR). Added to weight pointers inside the sequential time loop
// so the fragment-load addresses are loop-variant: LICM cannot hoist 1024
// VGPRs of loop-invariant B-fragments (which previously spilled to scratch).
// The pointer SSA chain stays intact, so address-space inference still emits
// global_load_b128 (not flat_load) for the weight stream.
__device__ __forceinline__ long opaque_zero() {
  long z = 0;
  asm volatile("" : "+s"(z));
  return z;
}

// A-fragment: 16x32 bf16, rows M in memory (row stride `ld` bf16, multiple of 8).
// Contiguous k order (same as load_b_frag): lanes 0-15 -> M=lane, K=0..15; lanes 16-31 -> K=16..31.
__device__ __forceinline__ v16bf load_a_frag(const bf16* base, int ld) {
  int lane = threadIdx.x & 31;
  const bf16* p = base + (lane & 15) * ld + ((lane >> 4) << 4);
  v16bf a;
#pragma unroll
  for (int e = 0; e < 8; ++e) { a[e] = p[e]; a[e + 8] = p[e + 8]; }
  return a;
}
// B-fragment: 32x16 bf16 taken from an N-major image BT[N][K] (stride `ld` bf16).
// lanes 0-15 -> N=lane, K=0..15; lanes 16-31 -> K=16..31.
__device__ __forceinline__ v16bf load_b_frag(const bf16* baseT, int ld) {
  int lane = threadIdx.x & 31;
  const bf16* p = baseT + (long)(lane & 15) * ld + ((lane >> 4) << 4);
  v16bf b;
#pragma unroll
  for (int e = 0; e < 16; ++e) b[e] = p[e];
  return b;
}
__device__ __forceinline__ v8f wmma_bf16(v16bf a, v16bf b, v8f c) {
  return __builtin_amdgcn_wmma_f32_16x16x32_bf16(false, a, false, b, (short)0, c, false, false);
}

// ---------------------------------------------------------------------------
// Generic tiled WMMA GEMM: C = epi(A[nr x K] @ W[K x nc] + bias) (opt rowscale/acc)
// Block = 256 threads (8 waves), tile 64x64, K-step 32.
// ---------------------------------------------------------------------------
enum { EPI_NONE = 0, EPI_SIGMOID, EPI_EXP, EPI_RELU, EPI_SOFTPLUS };
#define LDS_LD 40

template <int EPI, bool ACC, bool RSCALE>
__global__ __launch_bounds__(256) void gemm_kernel(
    const float* __restrict__ A, const float* __restrict__ W,
    const float* __restrict__ bias, float* __restrict__ C,
    int nrows, int K, int ncols, int lda, int ldw, int ldc,
    const float* __restrict__ rowscale, int rs_stride) {
  __shared__ __attribute__((aligned(16))) bf16 sA[64 * LDS_LD];
  __shared__ __attribute__((aligned(16))) bf16 sB[64 * LDS_LD];
  int tid = threadIdx.x;
  int wave = tid >> 5, lane = tid & 31;
  int row0 = blockIdx.y * 64;
  int col0 = blockIdx.x * 64;
  int t0 = wave * 2, t1 = t0 + 1;
  int rt0 = t0 >> 2, ct0 = t0 & 3, rt1 = t1 >> 2, ct1 = t1 & 3;
  v8f acc0 = {}, acc1 = {};

  for (int k0 = 0; k0 < K; k0 += 32) {
    __syncthreads();
#pragma unroll
    for (int i = 0; i < 8; ++i) {   // A tile 64x32, coalesced along k
      int e = tid + i * 256;
      int r = e >> 5, kk = e & 31;
      int gr = row0 + r, gk = k0 + kk;
      float v = (gr < nrows && gk < K) ? A[(long)gr * lda + gk] : 0.f;
      sA[r * LDS_LD + kk] = f2bf(v);
    }
#pragma unroll
    for (int i = 0; i < 8; ++i) {   // W tile 32x64 -> store N-major
      int e = tid + i * 256;
      int kk = e >> 6, n = e & 63;
      int gk = k0 + kk, gn = col0 + n;
      float v = (gk < K && gn < ncols) ? W[(long)gk * ldw + gn] : 0.f;
      sB[n * LDS_LD + kk] = f2bf(v);
    }
    if (k0 + 32 < K) {              // emits global_prefetch_b8
      __builtin_prefetch(&A[(long)(row0 + (tid >> 2)) * lda + k0 + 32], 0, 1);
      __builtin_prefetch(&W[(long)(k0 + 32 + (tid & 31)) * ldw + col0], 0, 1);
    }
    __syncthreads();
    v16bf a0 = load_a_frag(sA + rt0 * 16 * LDS_LD, LDS_LD);
    v16bf b0 = load_b_frag(sB + ct0 * 16 * LDS_LD, LDS_LD);
    acc0 = wmma_bf16(a0, b0, acc0);
    v16bf a1 = load_a_frag(sA + rt1 * 16 * LDS_LD, LDS_LD);
    v16bf b1 = load_b_frag(sB + ct1 * 16 * LDS_LD, LDS_LD);
    acc1 = wmma_bf16(a1, b1, acc1);
  }
#pragma unroll
  for (int ti = 0; ti < 2; ++ti) {
    v8f acc = ti ? acc1 : acc0;
    int rt = ti ? rt1 : rt0, ct = ti ? ct1 : ct0;
#pragma unroll
    for (int v = 0; v < 8; ++v) {
      int m = row0 + rt * 16 + v + ((lane >> 4) << 3);
      int n = col0 + ct * 16 + (lane & 15);
      if (m < nrows && n < ncols) {
        float val = acc[v];
        if (bias) val += bias[n];
        if (EPI == EPI_SIGMOID) val = sigmoidf_(val);
        else if (EPI == EPI_EXP) val = __expf(val);
        else if (EPI == EPI_RELU) val = fmaxf(val, 0.f);
        else if (EPI == EPI_SOFTPLUS) val = (val > 20.f) ? val : log1pf(__expf(val));
        if (RSCALE) val *= rowscale[(long)m * rs_stride];
        float* dst = &C[(long)m * ldc + n];
        if (ACC) *dst += val; else *dst = val;
      }
    }
  }
}

// ---------------------------------------------------------------------------
// 32x256 = (32xK bf16 in LDS) @ WT[256][K] (bf16, L2-resident) helper
// ---------------------------------------------------------------------------
#define HBF_LD 264
__device__ __forceinline__ v8f tile_gemm32(const bf16* hbf, const bf16* WT, int K,
                                           int rt, int ct) {
  v8f acc = {};
#pragma unroll
  for (int kk = 0; kk < K; kk += 32) {
    v16bf a = load_a_frag(hbf + rt * 16 * HBF_LD + kk, HBF_LD);
    v16bf b = load_b_frag(WT + (long)(ct * 16) * K + kk, K);
    acc = wmma_bf16(a, b, acc);
  }
  return acc;
}
__device__ __forceinline__ void gemm32x256(const bf16* hbf, const bf16* WT, float* gout) {
  int tid = threadIdx.x, wave = tid >> 5, lane = tid & 31;
#pragma unroll
  for (int i = 0; i < 4; ++i) {
    int tt = wave * 4 + i;
    int rt = tt >> 4, ct = tt & 15;
    v8f acc = tile_gemm32(hbf, WT, 256, rt, ct);
#pragma unroll
    for (int v = 0; v < 8; ++v) {
      int m = rt * 16 + v + ((lane >> 4) << 3);
      int n = ct * 16 + (lane & 15);
      gout[m * 256 + n] = acc[v];
    }
  }
}

// ---------------------------------------------------------------------------
// Small utility kernels
// ---------------------------------------------------------------------------
__global__ void transpose_bf16_kernel(const float* __restrict__ W, bf16* __restrict__ WT,
                                      int K, int N) {
  int i = blockIdx.x * blockDim.x + threadIdx.x;
  if (i < K * N) { int k = i / N, n = i % N; WT[(long)n * K + k] = f2bf(W[i]); }
}
__global__ void negexp_kernel(const float* in, float* out, int n) {
  int i = blockIdx.x * blockDim.x + threadIdx.x;
  if (i < n) out[i] = -__expf(in[i]);
}
__global__ void addvec_kernel(const float* a, const float* b, float* out, int n) {
  int i = blockIdx.x * blockDim.x + threadIdx.x;
  if (i < n) out[i] = a[i] + b[i];
}
__global__ void sens_combine_kernel(const float* a, const float* sg, const float* ex,
                                    float* out, int n) {
  int i = blockIdx.x * blockDim.x + threadIdx.x;
  if (i < n) out[i] = a[i] * sigmoidf_(sg[i]) * __expf(ex[i]);
}
__global__ void softmax8_kernel(float* w, int rows) {
  int r = blockIdx.x * blockDim.x + threadIdx.x;
  if (r >= rows) return;
  float l[8], mx = -1e30f;
#pragma unroll
  for (int j = 0; j < 8; ++j) { l[j] = w[r * 8 + j]; mx = fmaxf(mx, l[j]); }
  float s = 0.f;
#pragma unroll
  for (int j = 0; j < 8; ++j) { l[j] = __expf(l[j] - mx); s += l[j]; }
  float inv = 1.f / s;
#pragma unroll
  for (int j = 0; j < 8; ++j) w[r * 8 + j] = l[j] * inv;
}
__global__ void mem_init_kernel(const float* memory, float* mem) {
  int i = blockIdx.x * blockDim.x + threadIdx.x;
  if (i < BATCH * MROWS * HDIM) mem[i] = memory[i & (MROWS * HDIM - 1)];
}
__global__ void emb_gather_kernel(const int* tok, const float* emb, float* embx) {
  int i = blockIdx.x * blockDim.x + threadIdx.x;
  if (i >= BATCH * TLEN_ * EMBD) return;
  int e = i & (EMBD - 1); int bt = i >> 7;
  embx[i] = emb[(long)tok[bt] * EMBD + e];
}
__global__ void last_add_kernel(const float* mos, const float* fuse, float* last) {
  int i = blockIdx.x * blockDim.x + threadIdx.x;
  if (i >= BATCH * HDIM) return;
  int b = i >> 8, h = i & 255;
  last[i] = mos[((long)b * S_LEN + (S_LEN - 1)) * HDIM + h] + fuse[i];
}
__global__ void out_kernel(const float* o1, const float* w, const float* b, float* out) {
  int i = blockIdx.x * blockDim.x + threadIdx.x;
  if (i >= BATCH) return;
  float acc = b[0];
  for (int k = 0; k < 128; ++k) acc += o1[i * 128 + k] * w[k];
  out[i] = acc;
}

// ---------------------------------------------------------------------------
// LTC scan: one persistent workgroup, WMMA per step for h @ {tcw_h, iw, imu, isig}
// ---------------------------------------------------------------------------
__global__ __launch_bounds__(256, 1) void ltc_scan_kernel(
    const float* __restrict__ sens, const float* __restrict__ taux,
    const bf16* __restrict__ tcwhT_p, const bf16* __restrict__ iwT_p,
    const bf16* __restrict__ imuT_p, const bf16* __restrict__ isigT_p,
    const float* __restrict__ iw_b, const float* __restrict__ imu_b,
    const float* __restrict__ isig_b, float* __restrict__ hs) {
  extern __shared__ char smem[];
  float* h = (float*)smem;             // 32x256
  float* inter = h + BATCH * HDIM;     // 32x256
  float* gout = inter + BATCH * HDIM;  // 32x256
  bf16* hbf = (bf16*)(gout + BATCH * HDIM);  // 32 x HBF_LD
  int tid = threadIdx.x;
  for (int i = tid; i < BATCH * HDIM; i += 256) h[i] = 0.f;
  __syncthreads();
  for (int t = 0; t < S_LEN; ++t) {
    // loop-variant weight pointers: per-step fragments stream from L2
    long z = opaque_zero();
    const bf16* tcwhT = tcwhT_p + z;
    const bf16* iwT = iwT_p + z;
    const bf16* imuT = imuT_p + z;
    const bf16* isigT = isigT_p + z;
    for (int i = tid; i < BATCH * HDIM; i += 256)
      hbf[(i >> 8) * HBF_LD + (i & 255)] = f2bf(h[i]);
    __syncthreads();
    gemm32x256(hbf, iwT, gout);  __syncthreads();
    for (int i = tid; i < BATCH * HDIM; i += 256) inter[i] = gout[i] + iw_b[i & 255];
    __syncthreads();
    gemm32x256(hbf, imuT, gout); __syncthreads();
    for (int i = tid; i < BATCH * HDIM; i += 256) inter[i] *= sigmoidf_(gout[i] + imu_b[i & 255]);
    __syncthreads();
    gemm32x256(hbf, isigT, gout); __syncthreads();
    for (int i = tid; i < BATCH * HDIM; i += 256) inter[i] *= __expf(gout[i] + isig_b[i & 255]);
    __syncthreads();
    gemm32x256(hbf, tcwhT, gout); __syncthreads();
    for (int i = tid; i < BATCH * HDIM; i += 256) {
      int b = i >> 8, hh = i & 255;
      long n = (long)b * S_LEN + t;
      float tau = sigmoidf_(taux[n * HDIM + hh] + gout[i]) + 0.1f;
      float hv = h[i];
      hv = hv + 0.1f * (sens[n * HDIM + hh] + inter[i] - hv) / tau;
      h[i] = hv;
      hs[n * HDIM + hh] = hv;
    }
    __syncthreads();
  }
}

// ---------------------------------------------------------------------------
// SSM: Ad precompute (parallel), sdot, then trivial scan
// ---------------------------------------------------------------------------
__global__ void ssm_ad_kernel(const float* __restrict__ Aneg,
                              const float* __restrict__ dts, float* __restrict__ Ad) {
  int i = blockIdx.x * blockDim.x + threadIdx.x;
  if (i >= S_LEN * BATCH * NSTATE) return;
  int s = i & (NSTATE - 1); int b = (i >> 7) & 31; int t = i >> 12;
  const float* dtp = dts + ((long)b * S_LEN + t) * HDIM;
  float acc = 0.f;
  for (int hh = 0; hh < HDIM; ++hh) acc += __expf(Aneg[hh * NSTATE + s] * dtp[hh]);
  Ad[i] = acc;
}
__global__ void ssm_sdot_kernel(const float* __restrict__ dts,
                                const float* __restrict__ x, float* __restrict__ sdot) {
  int n = blockIdx.x * blockDim.x + threadIdx.x;
  if (n >= BATCH * S_LEN) return;
  float acc = 0.f;
  for (int hh = 0; hh < HDIM; ++hh) acc += dts[(long)n * HDIM + hh] * x[(long)n * HDIM + hh];
  sdot[n] = acc;
}
__global__ __launch_bounds__(256) void ssm_scan_kernel(
    const float* __restrict__ Ad, const float* __restrict__ xdbl,
    const float* __restrict__ sdot, const float* __restrict__ Dp,
    const float* __restrict__ x, float* __restrict__ y) {
  __shared__ float hsm[BATCH * NSTATE];
  __shared__ float dot[BATCH * 8];
  int tid = threadIdx.x;
  for (int i = tid; i < BATCH * NSTATE; i += 256) hsm[i] = 0.f;
  __syncthreads();
  const int LDX = DTRK + 2 * NSTATE;  // 272
  for (int t = 0; t < S_LEN; ++t) {
    for (int i = tid; i < BATCH * NSTATE; i += 256) {
      int b = i >> 7, s = i & (NSTATE - 1);
      long n = (long)b * S_LEN + t;
      float Bm = xdbl[n * LDX + DTRK + s];
      hsm[i] = Ad[((long)t * BATCH + b) * NSTATE + s] * hsm[i] + Bm * sdot[n];
    }
    __syncthreads();
    { int b = tid >> 3, l8 = tid & 7;
      long n = (long)b * S_LEN + t;
      float p = 0.f;
      for (int s = l8 * 16; s < l8 * 16 + 16; ++s)
        p += xdbl[n * LDX + DTRK + NSTATE + s] * hsm[b * NSTATE + s];
      dot[b * 8 + l8] = p; }
    __syncthreads();
    if (tid < BATCH) {
      float s = 0.f;
      for (int j = 0; j < 8; ++j) s += dot[tid * 8 + j];
      dot[tid * 8] = s;
    }
    __syncthreads();
    for (int i = tid; i < BATCH * HDIM; i += 256) {
      int b = i >> 8, hh = i & 255;
      long n = (long)b * S_LEN + t;
      y[n * HDIM + hh] = dot[b * 8] + Dp[hh] * x[n * HDIM + hh];
    }
    __syncthreads();
  }
}

// ---------------------------------------------------------------------------
// Memory-attention per-step fused kernel (1 block per batch element)
// ---------------------------------------------------------------------------
__global__ __launch_bounds__(256) void mem_attn_kernel(
    const float* __restrict__ qs, const float* __restrict__ xin,
    const float* __restrict__ kbuf, const float* __restrict__ vbuf,
    const float* __restrict__ ug_w, const float* __restrict__ ug_b,
    float* __restrict__ mem, float* __restrict__ mos, int t) {
  __shared__ float attn[MROWS];
  __shared__ float red[256];
  __shared__ float qsm[HDIM];
  __shared__ float xsm[HDIM];
  __shared__ float mo[HDIM];
  __shared__ float usm[HDIM];
  __shared__ int sel[MROWS];
  int b = blockIdx.x, tid = threadIdx.x;
  long n = (long)b * S_LEN + t;
  qsm[tid] = qs[n * HDIM + tid];
  xsm[tid] = xin[n * HDIM + tid];
  __syncthreads();
  { int m = tid >> 2, l4 = tid & 3;
    const float* kr = kbuf + ((long)b * MROWS + m) * HDIM;
    float p = 0.f;
    for (int hh = l4 * 64; hh < l4 * 64 + 64; ++hh) p += qsm[hh] * kr[hh];
    red[tid] = p; }
  __syncthreads();
  if (tid < MROWS)
    attn[tid] = (red[tid * 4] + red[tid * 4 + 1] + red[tid * 4 + 2] + red[tid * 4 + 3]) * 0.0625f;
  __syncthreads();
  if (tid == 0) {
    float mx = -1e30f;
    for (int m = 0; m < MROWS; ++m) mx = fmaxf(mx, attn[m]);
    float sm = 0.f;
    for (int m = 0; m < MROWS; ++m) { attn[m] = __expf(attn[m] - mx); sm += attn[m]; }
    float inv = 1.f / sm;
    for (int m = 0; m < MROWS; ++m) attn[m] *= inv;
  }
  __syncthreads();
  if (tid < MROWS) {  // top-8 by rank (stable tie-break = top_k semantics)
    int cnt = 0; float a = attn[tid];
    for (int j = 0; j < MROWS; ++j) {
      float aj = attn[j];
      if (aj > a || (aj == a && j < tid)) ++cnt;
    }
    sel[tid] = (cnt < 8) ? 1 : 0;
  }
  { float p = 0.f;
    const float* vb = vbuf + (long)b * MROWS * HDIM;
    for (int m = 0; m < MROWS; ++m) p += attn[m] * vb[m * HDIM + tid];
    mo[tid] = p;
    mos[n * HDIM + tid] = p; }
  __syncthreads();
  { float p = ug_b[tid];
    for (int k = 0; k < HDIM; ++k) p += xsm[k] * ug_w[(long)k * HDIM + tid];
    for (int k = 0; k < HDIM; ++k) p += mo[k] * ug_w[(long)(HDIM + k) * HDIM + tid];
    usm[tid] = sigmoidf_(p); }
  __syncthreads();
  float u = usm[tid], xv = xsm[tid];
  for (int m = 0; m < MROWS; ++m) {
    if (sel[m]) {
      float* mp = mem + ((long)b * MROWS + m) * HDIM + tid;
      *mp = (1.f - u) * (*mp) + u * xv;
    }
  }
}

// ---------------------------------------------------------------------------
// LSTM scan: persistent workgroup, WMMA for h @ whh per gate per step
// ---------------------------------------------------------------------------
__global__ __launch_bounds__(256, 1) void lstm_scan_kernel(
    const float* __restrict__ gih, const bf16* __restrict__ whhT_p,
    float* __restrict__ hT) {
  extern __shared__ char smem[];
  float* h = (float*)smem;            // 32x256
  float* c = h + BATCH * HDIM;        // 32x256
  float* si = c + BATCH * HDIM;       // 32x256
  bf16* hbf = (bf16*)(si + BATCH * HDIM);
  int tid = threadIdx.x, wave = tid >> 5, lane = tid & 31;
  for (int i = tid; i < BATCH * HDIM; i += 256) { h[i] = 0.f; c[i] = 0.f; }
  __syncthreads();
  for (int t = 0; t < TLEN_; ++t) {
    const bf16* whhT = whhT_p + opaque_zero();  // keep weight stream in-loop
    for (int i = tid; i < BATCH * HDIM; i += 256)
      hbf[(i >> 8) * HBF_LD + (i & 255)] = f2bf(h[i]);
    __syncthreads();
#pragma unroll
    for (int gsel = 0; gsel < 4; ++gsel) {
      int gate = (gsel == 0) ? 1 : (gsel == 1) ? 0 : (gsel == 2) ? 2 : 3;  // f,i,g,o
      const bf16* WTg = whhT + (long)gate * HDIM * HDIM;
#pragma unroll
      for (int i = 0; i < 4; ++i) {
        int tt = wave * 4 + i;
        int rt = tt >> 4, ct = tt & 15;
        v8f acc = tile_gemm32(hbf, WTg, HDIM, rt, ct);
#pragma unroll
        for (int v = 0; v < 8; ++v) {
          int m = rt * 16 + v + ((lane >> 4) << 3);
          int ncol = ct * 16 + (lane & 15);
          float g = acc[v] + gih[((long)(m * TLEN_ + t)) * (4 * HDIM) + gate * HDIM + ncol];
          int idx = m * HDIM + ncol;
          if (gate == 1)      c[idx] = sigmoidf_(g) * c[idx];
          else if (gate == 0) si[idx] = sigmoidf_(g);
          else if (gate == 2) c[idx] += si[idx] * tanhf(g);
          else                h[idx] = sigmoidf_(g) * tanhf(c[idx]);
        }
      }
      __syncthreads();
    }
  }
  for (int i = tid; i < BATCH * HDIM; i += 256) hT[i] = h[i];
}

// ---------------------------------------------------------------------------
// Host side
// ---------------------------------------------------------------------------
enum {
  IN_PRICE = 0, IN_TOK, P_IN_W, P_IN_B, P_SW_W, P_SW_B, P_SMU_W, P_SMU_B,
  P_SSIG_W, P_SSIG_B, P_IW_W, P_IW_B, P_IMU_W, P_IMU_B, P_ISIG_W, P_ISIG_B,
  P_TC_W, P_TC_B, P_G1_W, P_G1_B, P_G2_W, P_G2_B, P_SLOPES, P_INTER,
  P_XPROJ_W, P_XPROJ_B, P_DTPROJ_W, P_DTPROJ_B, P_A_LOG, P_DP, P_MEMORY,
  P_Q_W, P_Q_B, P_K_W, P_K_B, P_V_W, P_V_B, P_UG_W, P_UG_B, P_EMB,
  P_LSTM_WIH, P_LSTM_WHH, P_LSTM_BIH, P_LSTM_BHH, P_TXT_W, P_TXT_B,
  P_MHA_IN_W, P_MHA_IN_B, P_MHA_OUT_W, P_MHA_OUT_B, P_O1_W, P_O1_B, P_O2_W, P_O2_B
};

// workspace layout (float elements); total ~133 MB
#define NBS (8192L)
static const long F_X0   = 0;
static const long F_T1   = F_X0 + NBS * 256;
static const long F_T2   = F_T1 + NBS * 256;
static const long F_T3   = F_T2 + NBS * 256;
static const long F_SENS = F_T3 + NBS * 256;
static const long F_TAUX = F_SENS + NBS * 256;
static const long F_HS   = F_TAUX + NBS * 256;
static const long F_G1   = F_HS + NBS * 256;
static const long F_GW   = F_G1 + NBS * 64;
static const long F_MOE  = F_GW + NBS * 8;
static const long F_XDBL = F_MOE + NBS * 256;
static const long F_DTS  = F_XDBL + NBS * 272;
static const long F_A    = F_DTS + NBS * 256;
static const long F_AD   = F_A + 256L * 128;
static const long F_SDOT = F_AD + 256L * 32 * 128;
static const long F_Y    = F_SDOT + NBS;
static const long F_QS   = F_Y + NBS * 256;
static const long F_MEM  = F_QS + NBS * 256;
static const long F_K    = F_MEM + 32L * 64 * 256;
static const long F_V    = F_K + 32L * 64 * 256;
static const long F_MOS  = F_V + 32L * 64 * 256;
static const long F_EMBX = F_MOS + NBS * 256;
static const long F_GIH  = F_EMBX + 32L * 64 * 128;
static const long F_HT   = F_GIH + 2048L * 1024;
static const long F_TXT  = F_HT + 32L * 256;
static const long F_VH   = F_TXT + 32L * 256;
static const long F_FUSE = F_VH + 32L * 256;
static const long F_LAST = F_FUSE + 32L * 256;
static const long F_O1   = F_LAST + 32L * 256;
static const long F_BSUM = F_O1 + 32L * 128;
static const long F_BF16 = F_BSUM + 1024;
// bf16-element offsets within the bf16 region
static const long H_TCWH = 0;
static const long H_IW   = H_TCWH + 256L * 256;
static const long H_IMU  = H_IW + 256L * 256;
static const long H_ISIG = H_IMU + 256L * 256;
static const long H_WHH  = H_ISIG + 256L * 256;

static void launch_gemm(hipStream_t st, int epi, bool acc, bool rscale,
                        const float* A, const float* W, const float* bias, float* C,
                        int nr, int K, int nc, int lda, int ldw, int ldc,
                        const float* rs, int rss) {
  dim3 grid((nc + 63) / 64, (nr + 63) / 64), block(256);
#define LG(E, AC, RS) gemm_kernel<E, AC, RS><<<grid, block, 0, st>>>(A, W, bias, C, nr, K, nc, lda, ldw, ldc, rs, rss)
  if (rscale) {
    if (acc) LG(EPI_NONE, true, true); else LG(EPI_NONE, false, true);
  } else {
    switch (epi) {
      case EPI_SIGMOID:  LG(EPI_SIGMOID, false, false); break;
      case EPI_EXP:      LG(EPI_EXP, false, false); break;
      case EPI_RELU:     LG(EPI_RELU, false, false); break;
      case EPI_SOFTPLUS: LG(EPI_SOFTPLUS, false, false); break;
      default:           LG(EPI_NONE, false, false); break;
    }
  }
#undef LG
}

extern "C" void kernel_launch(void* const* d_in, const int* in_sizes, int n_in,
                              void* d_out, int out_size, void* d_ws, size_t ws_size,
                              hipStream_t stream) {
  (void)in_sizes; (void)n_in; (void)out_size; (void)ws_size;
  const float* price = (const float*)d_in[IN_PRICE];
  const int* tok = (const int*)d_in[IN_TOK];
  auto P = [&](int i) { return (const float*)d_in[i]; };
  float* ws = (float*)d_ws;
  bf16* wsh = (bf16*)(ws + F_BF16);
  float* out = (float*)d_out;

  // --- prep: bf16-transposed recurrent weights, A = -exp(A_log), LSTM bias sum
  { int n = 256 * 256;
    transpose_bf16_kernel<<<(n + 255) / 256, 256, 0, stream>>>(P(P_TC_W) + 256 * 256, wsh + H_TCWH, 256, 256);
    transpose_bf16_kernel<<<(n + 255) / 256, 256, 0, stream>>>(P(P_IW_W),   wsh + H_IW,   256, 256);
    transpose_bf16_kernel<<<(n + 255) / 256, 256, 0, stream>>>(P(P_IMU_W),  wsh + H_IMU,  256, 256);
    transpose_bf16_kernel<<<(n + 255) / 256, 256, 0, stream>>>(P(P_ISIG_W), wsh + H_ISIG, 256, 256);
    int n2 = 256 * 1024;
    transpose_bf16_kernel<<<(n2 + 255) / 256, 256, 0, stream>>>(P(P_LSTM_WHH), wsh + H_WHH, 256, 1024);
    negexp_kernel<<<(256 * 128 + 255) / 256, 256, 0, stream>>>(P(P_A_LOG), ws + F_A, 256 * 128);
    addvec_kernel<<<4, 256, 0, stream>>>(P(P_LSTM_BIH), P(P_LSTM_BHH), ws + F_BSUM, 1024); }

  // --- input projection: x0 = price @ in_w + in_b
  launch_gemm(stream, EPI_NONE, false, false, price, P(P_IN_W), P(P_IN_B), ws + F_X0,
              8192, PDIM, HDIM, PDIM, HDIM, HDIM, nullptr, 0);
  // --- sensor branch
  launch_gemm(stream, EPI_NONE, false, false, ws + F_X0, P(P_SW_W),   P(P_SW_B),   ws + F_T1, 8192, 256, 256, 256, 256, 256, nullptr, 0);
  launch_gemm(stream, EPI_NONE, false, false, ws + F_X0, P(P_SMU_W),  P(P_SMU_B),  ws + F_T2, 8192, 256, 256, 256, 256, 256, nullptr, 0);
  launch_gemm(stream, EPI_NONE, false, false, ws + F_X0, P(P_SSIG_W), P(P_SSIG_B), ws + F_T3, 8192, 256, 256, 256, 256, 256, nullptr, 0);
  sens_combine_kernel<<<8192, 256, 0, stream>>>(ws + F_T1, ws + F_T2, ws + F_T3, ws + F_SENS, 8192 * 256);
  // --- taux = x0 @ tcw_x + tc_b
  launch_gemm(stream, EPI_NONE, false, false, ws + F_X0, P(P_TC_W), P(P_TC_B), ws + F_TAUX, 8192, 256, 256, 256, 256, 256, nullptr, 0);
  // --- LTC scan
  { size_t sm = (size_t)(3 * BATCH * HDIM) * 4 + (size_t)BATCH * HBF_LD * 2;
    ltc_scan_kernel<<<1, 256, sm, stream>>>(ws + F_SENS, ws + F_TAUX,
        wsh + H_TCWH, wsh + H_IW, wsh + H_IMU, wsh + H_ISIG,
        P(P_IW_B), P(P_IMU_B), P(P_ISIG_B), ws + F_HS); }
  // --- MoE gating + weighted pieces
  launch_gemm(stream, EPI_RELU, false, false, ws + F_HS, P(P_G1_W), P(P_G1_B), ws + F_G1, 8192, 256, 64, 256, 64, 64, nullptr, 0);
  launch_gemm(stream, EPI_NONE, false, false, ws + F_G1, P(P_G2_W), P(P_G2_B), ws + F_GW, 8192, 64, 8, 64, 8, 8, nullptr, 0);
  softmax8_kernel<<<32, 256, 0, stream>>>(ws + F_GW, 8192);
  for (int p = 0; p < NPCNT; ++p) {
    launch_gemm(stream, EPI_NONE, p > 0, true, ws + F_HS, P(P_SLOPES) + (long)p * 256 * 256,
                P(P_INTER) + (long)p * 256, ws + F_MOE, 8192, 256, 256, 256, 256, 256,
                ws + F_GW + p, 8);
  }
  // --- SSM projections
  launch_gemm(stream, EPI_NONE, false, false, ws + F_MOE, P(P_XPROJ_W), P(P_XPROJ_B), ws + F_XDBL, 8192, 256, 272, 256, 272, 272, nullptr, 0);
  launch_gemm(stream, EPI_SOFTPLUS, false, false, ws + F_XDBL, P(P_DTPROJ_W), P(P_DTPROJ_B), ws + F_DTS, 8192, 16, 256, 272, 256, 256, nullptr, 0);
  // --- SSM: hoisted Ad + sdot, then scan
  ssm_ad_kernel<<<(256 * 32 * 128 + 255) / 256, 256, 0, stream>>>(ws + F_A, ws + F_DTS, ws + F_AD);
  ssm_sdot_kernel<<<32, 256, 0, stream>>>(ws + F_DTS, ws + F_MOE, ws + F_SDOT);
  ssm_scan_kernel<<<1, 256, 0, stream>>>(ws + F_AD, ws + F_XDBL, ws + F_SDOT, P(P_DP), ws + F_MOE, ws + F_Y);
  // --- memory attention scan
  launch_gemm(stream, EPI_NONE, false, false, ws + F_Y, P(P_Q_W), P(P_Q_B), ws + F_QS, 8192, 256, 256, 256, 256, 256, nullptr, 0);
  mem_init_kernel<<<(32 * 64 * 256 + 255) / 256, 256, 0, stream>>>(P(P_MEMORY), ws + F_MEM);
  for (int t = 0; t < S_LEN; ++t) {
    launch_gemm(stream, EPI_NONE, false, false, ws + F_MEM, P(P_K_W), P(P_K_B), ws + F_K, 2048, 256, 256, 256, 256, 256, nullptr, 0);
    launch_gemm(stream, EPI_NONE, false, false, ws + F_MEM, P(P_V_W), P(P_V_B), ws + F_V, 2048, 256, 256, 256, 256, 256, nullptr, 0);
    mem_attn_kernel<<<32, 256, 0, stream>>>(ws + F_QS, ws + F_Y, ws + F_K, ws + F_V,
                                            P(P_UG_W), P(P_UG_B), ws + F_MEM, ws + F_MOS, t);
  }
  // --- LSTM over text
  emb_gather_kernel<<<(32 * 64 * 128 + 255) / 256, 256, 0, stream>>>(tok, P(P_EMB), ws + F_EMBX);
  launch_gemm(stream, EPI_NONE, false, false, ws + F_EMBX, P(P_LSTM_WIH), ws + F_BSUM, ws + F_GIH, 2048, 128, 1024, 128, 1024, 1024, nullptr, 0);
  { size_t sm = (size_t)(3 * BATCH * HDIM) * 4 + (size_t)BATCH * HBF_LD * 2;
    lstm_scan_kernel<<<1, 256, sm, stream>>>(ws + F_GIH, wsh + H_WHH, ws + F_HT); }
  // --- txt projection + collapsed MHA (K/V constant across keys -> attn uniform -> out = V)
  launch_gemm(stream, EPI_NONE, false, false, ws + F_HT, P(P_TXT_W), P(P_TXT_B), ws + F_TXT, 32, 256, 256, 256, 256, 256, nullptr, 0);
  launch_gemm(stream, EPI_NONE, false, false, ws + F_TXT, P(P_MHA_IN_W) + 512, P(P_MHA_IN_B) + 512, ws + F_VH, 32, 256, 256, 256, 768, 256, nullptr, 0);
  launch_gemm(stream, EPI_NONE, false, false, ws + F_VH, P(P_MHA_OUT_W), P(P_MHA_OUT_B), ws + F_FUSE, 32, 256, 256, 256, 256, 256, nullptr, 0);
  // --- output head on x[:, -1, :] + fused
  last_add_kernel<<<32, 256, 0, stream>>>(ws + F_MOS, ws + F_FUSE, ws + F_LAST);
  launch_gemm(stream, EPI_RELU, false, false, ws + F_LAST, P(P_O1_W), P(P_O1_B), ws + F_O1, 32, 256, 128, 256, 128, 128, nullptr, 0);
  out_kernel<<<1, 32, 0, stream>>>(ws + F_O1, P(P_O2_W), P(P_O2_B), out);
}
